// MoE_44573170598229
// MI455X (gfx1250) — compile-verified, matrix operand order checked by reading, not run on hardware
//
#include <hip/hip_runtime.h>
#include <hip/hip_bf16.h>

typedef __attribute__((ext_vector_type(16))) _Float16 v16h;
typedef __attribute__((ext_vector_type(8)))  _Float16 h8;
typedef __attribute__((ext_vector_type(8)))  float    v8f;
typedef __attribute__((ext_vector_type(4)))  float    f32x4;
typedef __attribute__((ext_vector_type(4)))  int      i32x4;

#if defined(__has_builtin)
#if __has_builtin(__builtin_amdgcn_global_load_async_to_lds_b128)
#define ASYNC_LDS 1
#endif
#if __has_builtin(__builtin_amdgcn_s_wait_asynccnt)
#define HAVE_WAIT_ASYNC 1
#endif
#endif

#ifdef ASYNC_LDS
typedef __attribute__((address_space(1))) i32x4 gas_i32x4;  // global
typedef __attribute__((address_space(3))) i32x4 las_i32x4;  // LDS
#endif

// ---------------------------------------------------------------------------
// f32 -> f16 flat conversion (one-time pass).
// ---------------------------------------------------------------------------
__global__ __launch_bounds__(256) void cvt_f32_to_f16_kernel(
    const float* __restrict__ src, _Float16* __restrict__ dst, int n)
{
    int i = (blockIdx.x * 256 + threadIdx.x) * 8;
    if (i >= n) return;
    f32x4 a = *(const f32x4*)(src + i);
    f32x4 b = *(const f32x4*)(src + i + 4);
    h8 o;
#pragma unroll
    for (int j = 0; j < 4; ++j) {
        o[j]     = (_Float16)a[j];
        o[4 + j] = (_Float16)b[j];
    }
    *(h8*)(dst + i) = o;
}

// Fragment load (A/B role) from f16 memory: two b128 loads, zero VALU.
__device__ __forceinline__ v16h load_frag_f16(const _Float16* __restrict__ p) {
    h8 lo = *(const h8*)(p);
    h8 hi = *(const h8*)(p + 16);
    v16h f;
#pragma unroll
    for (int i = 0; i < 8; ++i) { f[i] = lo[i]; f[8 + i] = hi[i]; }
    return f;
}

__device__ __forceinline__ float elu_f(float y) {
    return y > 0.0f ? y : (__expf(y) - 1.0f);
}

// Stage one 64B row chunk (32 halfs) global -> LDS.
__device__ __forceinline__ void stage_row(const _Float16* __restrict__ g,
                                          _Float16* l)
{
#ifdef ASYNC_LDS
#pragma unroll
    for (int i = 0; i < 4; ++i) {
        __builtin_amdgcn_global_load_async_to_lds_b128(
            (gas_i32x4*)(g + i * 8),
            (las_i32x4*)(l + i * 8), 0, 0);
    }
#else
    const h8* s = (const h8*)g;
    h8* d = (h8*)l;
    d[0] = s[0]; d[1] = s[1]; d[2] = s[2]; d[3] = s[3];
#endif
}

__device__ __forceinline__ void wait_stage() {
#ifdef ASYNC_LDS
#ifdef HAVE_WAIT_ASYNC
    __builtin_amdgcn_s_wait_asynccnt(0);
#else
    asm volatile("s_wait_asynccnt 0" ::: "memory");
#endif
#endif
}

// ---------------------------------------------------------------------------
// Y[M,N] = act(X @ W^T + bias); f16 out. One wave per 16x16 tile. (Gating.)
// ---------------------------------------------------------------------------
template <bool ELU>
__global__ __launch_bounds__(128) void gemm_kernel(
    const _Float16* __restrict__ X, const _Float16* __restrict__ W,
    const float* __restrict__ bias, _Float16* __restrict__ Y,
    int M, int N, int K)
{
    const int lane = threadIdx.x & 31;
    const int wave = threadIdx.x >> 5;
    const int tn   = blockIdx.x * 4 + wave;
    const int tm   = blockIdx.y;
    const int r    = lane & 15;
    const int hh   = lane >> 4;

    const _Float16* arow = X + (size_t)(tm * 16 + r) * K + hh * 8;
    const _Float16* brow = W + (size_t)(tn * 16 + r) * K + hh * 8;

    v8f acc = {};
    for (int k = 0; k < K; k += 32) {
        v16h a = load_frag_f16(arow + k);
        v16h b = load_frag_f16(brow + k);
        acc = __builtin_amdgcn_wmma_f32_16x16x32_f16(
            false, a, false, b, (short)0, acc, false, false);
    }

    const int ncol = tn * 16 + r;
    const float bv = bias[ncol];
    _Float16* yp = Y + (size_t)(tm * 16 + hh * 8) * N + ncol;
#pragma unroll
    for (int v = 0; v < 8; ++v) {
        float y = acc[v] + bv;
        if constexpr (ELU) y = elu_f(y);
        yp[(size_t)v * N] = (_Float16)y;
    }
}

// ---------------------------------------------------------------------------
// g = softmax over E=8 logits. One wave per batch row.
// ---------------------------------------------------------------------------
__global__ __launch_bounds__(256) void gate_softmax_kernel(
    const _Float16* __restrict__ Hin, const float* __restrict__ W2,
    const float* __restrict__ b2, float* __restrict__ g, int Bn, int K)
{
    const int lane = threadIdx.x & 31;
    const int wid  = threadIdx.x >> 5;
    const int b    = blockIdx.x * 8 + wid;
    if (b >= Bn) return;

    float logit[8];
#pragma unroll
    for (int e = 0; e < 8; ++e) {
        float s = 0.0f;
        for (int k = lane; k < K; k += 32)
            s += (float)Hin[(size_t)b * K + k] * W2[(size_t)e * K + k];
#pragma unroll
        for (int off = 16; off > 0; off >>= 1)
            s += __shfl_xor(s, off, 32);
        logit[e] = s + b2[e];
    }
    float mx = logit[0];
#pragma unroll
    for (int e = 1; e < 8; ++e) mx = fmaxf(mx, logit[e]);
    float ex[8];
    float den = 0.0f;
#pragma unroll
    for (int e = 0; e < 8; ++e) { ex[e] = __expf(logit[e] - mx); den += ex[e]; }
    if (lane == 0) {
        float inv = 1.0f / den;
#pragma unroll
        for (int e = 0; e < 8; ++e) g[(size_t)b * 8 + e] = ex[e] * inv;
    }
}

// ---------------------------------------------------------------------------
// MoE layer: Y[b,n] = act( sum_e g[b,e]*(X[b,:].alpha[e,n,:] + beta[e,n]) )
// Block = 4 waves sharing ONE N-tile (same B operand), covering 4 M-tiles.
// 8-expert B tile (8KB/k-step) staged in LDS, double-buffered; each wave
// holds 8 f32 accumulators (one per expert) and reuses its A fragment 8x.
// ---------------------------------------------------------------------------
template <bool ELU, bool F32OUT>
__global__ __launch_bounds__(128) void moe_kernel(
    const _Float16* __restrict__ X, const _Float16* __restrict__ alpha,
    const float* __restrict__ beta, const float* __restrict__ g,
    void* __restrict__ Yp, int M, int N, int K)
{
    constexpr int LROW = 40;  // padded row stride (halfs): 80B rows, 16B aligned
    __shared__ _Float16 Bs[2][8 * 16 * LROW];

    const int lane = threadIdx.x & 31;
    const int wave = threadIdx.x >> 5;
    const int tn   = blockIdx.x;              // one N tile per block
    const int tm   = blockIdx.y * 4 + wave;   // 4 M tiles per block
    const int r    = lane & 15;
    const int hh   = lane >> 4;

    // staging role: thread t covers (expert t>>4, row t&15) of the B tile
    const int se = threadIdx.x >> 4;
    const int sr = threadIdx.x & 15;
    const size_t estride = (size_t)N * K;
    const _Float16* sgsrc = alpha + (size_t)se * estride + (size_t)(tn * 16 + sr) * K;
    _Float16* sdst0 = &Bs[0][(se * 16 + sr) * LROW];
    _Float16* sdst1 = &Bs[1][(se * 16 + sr) * LROW];

    const _Float16* arow = X + (size_t)(tm * 16 + r) * K + hh * 8;

    v8f acc[8];
    v8f z = {};
#pragma unroll
    for (int e = 0; e < 8; ++e) acc[e] = z;

    stage_row(sgsrc, sdst0);  // prologue: k-chunk 0 -> buffer 0

    const int nk = K / 32;
    for (int kc = 0; kc < nk; ++kc) {
        const int cur = kc & 1;
        wait_stage();          // my async copies into Bs[cur] done
        __syncthreads();       // everyone's copies visible; prior reads done
        if (kc + 1 < nk)
            stage_row(sgsrc + (size_t)(kc + 1) * 32, cur ? sdst0 : sdst1);

        v16h a = load_frag_f16(arow + (size_t)kc * 32);
        const _Float16* bbase = &Bs[cur][r * LROW + hh * 8];
#pragma unroll
        for (int e = 0; e < 8; ++e) {
            const _Float16* bp = bbase + e * 16 * LROW;
            h8 lo = *(const h8*)bp;
            h8 hi = *(const h8*)(bp + 16);
            v16h b;
#pragma unroll
            for (int i = 0; i < 8; ++i) { b[i] = lo[i]; b[8 + i] = hi[i]; }
            acc[e] = __builtin_amdgcn_wmma_f32_16x16x32_f16(
                false, a, false, b, (short)0, acc[e], false, false);
        }
    }

    const int ncol = tn * 16 + r;
    float bb[8];
#pragma unroll
    for (int e = 0; e < 8; ++e) bb[e] = beta[(size_t)e * N + ncol];

    const int row0 = tm * 16 + hh * 8;
#pragma unroll
    for (int v = 0; v < 8; ++v) {
        const float* gr = g + (size_t)(row0 + v) * 8;
        float y = 0.0f;
#pragma unroll
        for (int e = 0; e < 8; ++e) y += gr[e] * (acc[e][v] + bb[e]);
        if constexpr (ELU) y = elu_f(y);
        if constexpr (F32OUT)
            ((float*)Yp)[(size_t)(row0 + v) * N + ncol] = y;
        else
            ((_Float16*)Yp)[(size_t)(row0 + v) * N + ncol] = (_Float16)y;
    }
}

extern "C" void kernel_launch(void* const* d_in, const int* in_sizes, int n_in,
                              void* d_out, int out_size, void* d_ws, size_t ws_size,
                              hipStream_t stream) {
    (void)in_sizes; (void)n_in; (void)out_size; (void)ws_size;

    const float* x      = (const float*)d_in[0];
    const float* gw0    = (const float*)d_in[1];
    const float* gb0    = (const float*)d_in[2];
    const float* gw1    = (const float*)d_in[3];
    const float* gb1    = (const float*)d_in[4];
    const float* gw2    = (const float*)d_in[5];
    const float* gb2    = (const float*)d_in[6];
    const float* alpha0 = (const float*)d_in[7];
    const float* beta0  = (const float*)d_in[8];
    const float* alpha1 = (const float*)d_in[9];
    const float* beta1  = (const float*)d_in[10];
    const float* alpha2 = (const float*)d_in[11];
    const float* beta2  = (const float*)d_in[12];
    float* out = (float*)d_out;

    const int Bn = 512, F = 512, Hn = 512, O = 512, E = 8;

    _Float16* w = (_Float16*)d_ws;
    _Float16* x16  = w; w += (size_t)Bn * F;
    _Float16* w016 = w; w += (size_t)Hn * F;
    _Float16* w116 = w; w += (size_t)Hn * Hn;
    _Float16* a016 = w; w += (size_t)E * Hn * F;
    _Float16* a116 = w; w += (size_t)E * Hn * Hn;
    _Float16* a216 = w; w += (size_t)E * O * Hn;
    _Float16* h016 = w; w += (size_t)Bn * Hn;
    _Float16* h116 = w; w += (size_t)Bn * Hn;
    _Float16* H116 = w; w += (size_t)Bn * Hn;
    _Float16* H216 = w; w += (size_t)Bn * Hn;
    float* gg = (float*)w;  // [B, 8]

    auto cvt = [&](const float* s, _Float16* d, int n) {
        cvt_f32_to_f16_kernel<<<dim3((n / 8 + 255) / 256), dim3(256), 0, stream>>>(s, d, n);
    };
    cvt(x,      x16,  Bn * F);
    cvt(gw0,    w016, Hn * F);
    cvt(gw1,    w116, Hn * Hn);
    cvt(alpha0, a016, E * Hn * F);
    cvt(alpha1, a116, E * Hn * Hn);
    cvt(alpha2, a216, E * O * Hn);

    dim3 blk(128);
    dim3 grdG(Hn / 64, Bn / 16);
    dim3 grdM(Hn / 16, Bn / 64);

    gemm_kernel<true><<<grdG, blk, 0, stream>>>(x16,  w016, gb0, h016, Bn, Hn, F);
    gemm_kernel<true><<<grdG, blk, 0, stream>>>(h016, w116, gb1, h116, Bn, Hn, Hn);
    gate_softmax_kernel<<<dim3(Bn / 8), dim3(256), 0, stream>>>(h116, gw2, gb2, gg, Bn, Hn);
    moe_kernel<true,  false><<<grdM, blk, 0, stream>>>(x16,  a016, beta0, gg, H116, Bn, Hn, F);
    moe_kernel<true,  false><<<grdM, blk, 0, stream>>>(H116, a116, beta1, gg, H216, Bn, Hn, Hn);
    moe_kernel<false, true ><<<dim3(O / 16, Bn / 64), blk, 0, stream>>>(
        H216, a216, beta2, gg, out, Bn, O, Hn);
}
